// DeepHisCoM_7017976562218
// MI455X (gfx1250) — matrix operand order repaired and checked by reading, not compiled, not run on hardware
//
#include <hip/hip_runtime.h>

#define NPATH 128
#define NVAR  128
#define NHID  64
#define BATCH 16384
#define BN_EPS 1e-5f
#define XSTRIDE 132   // 128 + 4 bf16 pad -> conflict-mitigating row stride

typedef __attribute__((ext_vector_type(16))) __bf16 v16bf;
typedef __attribute__((ext_vector_type(2)))  __bf16 v2bf;
typedef __attribute__((ext_vector_type(8)))  float  v8f;

union Frag16 { v16bf v; unsigned int u[8]; };

// Native bf16 convert (RNE fptrunc) -> lets clang emit v_cvt_(pk_)bf16_f32
__device__ __forceinline__ unsigned int pack_bf16(float a, float b) {
    v2bf t; t[0] = (__bf16)a; t[1] = (__bf16)b;
    return __builtin_bit_cast(unsigned int, t);
}
// leaky(z) == max(z, 0.2z) for all reals: mul+max, no cmp/cndmask
__device__ __forceinline__ float leaky(float z) { return fmaxf(z, 0.2f * z); }

// ---------------------------------------------------------------------------
// Kernel A: per-pathway 2-layer MLP, fused. p_ws stored column-major [P][B].
// ---------------------------------------------------------------------------
__global__ __launch_bounds__(128) void pathway_mlp_kernel(
    const float* __restrict__ x, const float* __restrict__ W1,
    const float* __restrict__ W2, float* __restrict__ p_ws)
{
    __shared__ __align__(16) unsigned short lds_w1[NHID * XSTRIDE];
    __shared__ __align__(16) unsigned short lds_x[4][16 * XSTRIDE];

    const int path = blockIdx.x;
    const int tid  = threadIdx.x;
    const int wave = tid >> 5;
    const int lane = tid & 31;

    // Stage W1[path] (64x128 fp32) -> LDS bf16, pairwise packed, coalesced.
    const float* w1g = W1 + (size_t)path * (NHID * NVAR);
    for (int i = tid; i < (NHID * NVAR) / 2; i += 128) {
        int w = i >> 6, v2 = i & 63;
        const float2 q = *(const float2*)(w1g + 2 * i);
        *(unsigned int*)(lds_w1 + w * XSTRIDE + 2 * v2) = pack_bf16(q.x, q.y);
    }
    __syncthreads();

    const int nl   = lane & 15;
    const int half = lane >> 4;
    const int kg   = half * 16;

    // Build all B fragments (K x N = 128 x 64 -> 4 ntiles x 4 ksteps), kept in VGPRs.
    // B layout: lane = N column (nl), low lanes K=0..7/16..23, high lanes +8 per kstep.
    Frag16 bfr[16];
#pragma unroll
    for (int nt = 0; nt < 4; ++nt) {
        const unsigned short* wrow = lds_w1 + (nl + 16 * nt) * XSTRIDE;
#pragma unroll
        for (int s = 0; s < 4; ++s)
#pragma unroll
            for (int j = 0; j < 8; ++j)
                bfr[nt * 4 + s].u[j] =
                    *(const unsigned int*)(wrow + 32 * s + (j < 4 ? 2 * j : 16 + 2 * (j - 4)) + 8 * half);
    }

    float w2v[4];
#pragma unroll
    for (int nt = 0; nt < 4; ++nt) w2v[nt] = W2[path * NHID + nl + 16 * nt];

    unsigned short* myx = lds_x[wave];
    const unsigned short* arow = myx + nl * XSTRIDE;   // this lane's A row (m = nl)

    for (int t = blockIdx.y * 4 + wave; t < BATCH / 16; t += gridDim.y * 4) {
        const int rowBase = t * 16;
        // Stage 16x128 x-tile -> LDS bf16 (coalesced float4 loads, packed cvt).
        const float* xg = x + (size_t)rowBase * (NPATH * NVAR) + path * NVAR;
#pragma unroll
        for (int i = 0; i < 16; ++i) {
            int f = lane + 32 * i;            // 0..511
            int r = f >> 5, c4 = f & 31;
            const float4 q = *(const float4*)(xg + (size_t)r * (NPATH * NVAR) + c4 * 4);
            *(unsigned int*)(myx + r * XSTRIDE + c4 * 4)     = pack_bf16(q.x, q.y);
            *(unsigned int*)(myx + r * XSTRIDE + c4 * 4 + 2) = pack_bf16(q.z, q.w);
        }
        // (per-wave LDS buffer: compiler inserts s_wait_dscnt for same-wave RAW)

        v8f acc0 = {}, acc1 = {}, acc2 = {}, acc3 = {};
        // A fragment, 16-bit A 16x32 layout: lanes 0-15 K=0..7/16..23, lanes 16-31 +8.
        Frag16 acur, anxt;
#pragma unroll
        for (int j = 0; j < 8; ++j) {
            int k0 = (j < 4 ? 2 * j : 16 + 2 * (j - 4)) + 8 * half;
            acur.u[j] = *(const unsigned int*)(arow + k0);
        }
#pragma unroll
        for (int s = 0; s < 4; ++s) {
            if (s < 3) {   // prefetch next k-step's A fragment above this step's WMMAs
#pragma unroll
                for (int j = 0; j < 8; ++j) {
                    int k0 = (j < 4 ? 2 * j : 16 + 2 * (j - 4)) + 8 * half;
                    anxt.u[j] = *(const unsigned int*)(arow + 32 * (s + 1) + k0);
                }
            }
            acc0 = __builtin_amdgcn_wmma_f32_16x16x32_bf16(false, acur.v, false, bfr[0 * 4 + s].v, (short)0, acc0, false, false);
            acc1 = __builtin_amdgcn_wmma_f32_16x16x32_bf16(false, acur.v, false, bfr[1 * 4 + s].v, (short)0, acc1, false, false);
            acc2 = __builtin_amdgcn_wmma_f32_16x16x32_bf16(false, acur.v, false, bfr[2 * 4 + s].v, (short)0, acc2, false, false);
            acc3 = __builtin_amdgcn_wmma_f32_16x16x32_bf16(false, acur.v, false, bfr[3 * 4 + s].v, (short)0, acc3, false, false);
            if (s < 3) acur = anxt;
        }

        // Fused layer 2: p = leaky( leaky(h) . W2 ); reduce across 16-lane groups.
        float s8[8];
#pragma unroll
        for (int r = 0; r < 8; ++r) {
            float v0 = leaky(acc0[r]) * w2v[0] + leaky(acc1[r]) * w2v[1]
                     + leaky(acc2[r]) * w2v[2] + leaky(acc3[r]) * w2v[3];
#pragma unroll
            for (int msk = 1; msk < 16; msk <<= 1)
                v0 += __shfl_xor(v0, msk, 32);
            s8[r] = leaky(v0);
        }
        if (nl == 0) {   // lane 0 -> m=0..7, lane 16 -> m=8..15
            float* dst = p_ws + (size_t)path * BATCH + rowBase + 8 * half;
#pragma unroll
            for (int r = 0; r < 8; ++r) dst[r] = s8[r];
        }
    }
}

// ---------------------------------------------------------------------------
// Scratch stats region layout (floats): [0:128)=mean [128:256)=istd
// [256:384)=coef [384]=normsq-acc [385]=bias constant C
// ---------------------------------------------------------------------------
__global__ void init_kernel(float* stats) { stats[384] = 0.f; }

__global__ __launch_bounds__(256) void bn_stats_kernel(
    const float* __restrict__ p_ws, const float* __restrict__ gamma,
    const float* __restrict__ beta, float* __restrict__ stats)
{
    __shared__ float red[256], red2[256];
    const int path = blockIdx.x, t = threadIdx.x;
    const float* col = p_ws + (size_t)path * BATCH;
    float s = 0.f, s2 = 0.f;
    for (int i = t; i < BATCH; i += 256) { float v = col[i]; s += v; s2 += v * v; }
    red[t] = s; red2[t] = s2; __syncthreads();
    for (int o = 128; o > 0; o >>= 1) {
        if (t < o) { red[t] += red[t + o]; red2[t] += red2[t + o]; }
        __syncthreads();
    }
    if (t == 0) {
        float mean = red[0] / (float)BATCH;
        float var  = red2[0] / (float)BATCH - mean * mean;   // biased, as jnp.var
        if (var < 0.f) var = 0.f;
        float istd = rsqrtf(var + BN_EPS);
        stats[path]       = mean;
        stats[128 + path] = istd;
        // ||pn||^2 contribution (analytic): g^2*B*var/(var+eps) + B*beta^2
        float g = gamma[path], b = beta[path];
        atomicAdd(&stats[384], g * g * (float)BATCH * var * istd * istd
                               + (float)BATCH * b * b);
    }
}

__global__ __launch_bounds__(128) void coef_kernel(
    const float* __restrict__ Wd, const float* __restrict__ bd,
    const float* __restrict__ gamma, const float* __restrict__ beta,
    float* __restrict__ stats)
{
    __shared__ float red[128];
    __shared__ float invnorm_s;
    const int p = threadIdx.x;
    float mean = stats[p], istd = stats[128 + p];
    float wd = Wd[p], g = gamma[p], b = beta[p];
    red[p] = (b - mean * istd * g) * wd;
    __syncthreads();
    for (int o = 64; o > 0; o >>= 1) { if (p < o) red[p] += red[p + o]; __syncthreads(); }
    if (p == 0) {
        float invn = rsqrtf(stats[384]);
        invnorm_s  = invn;
        stats[385] = bd[0] + red[0] * invn;
    }
    __syncthreads();
    stats[256 + p] = wd * g * istd * invnorm_s;
}

__global__ __launch_bounds__(256) void out_kernel(
    const float* __restrict__ p_ws, const float* __restrict__ stats,
    float* __restrict__ out)
{
    __shared__ float coef[128];
    __shared__ float Cs;
    const int t = threadIdx.x;
    if (t < 128) coef[t] = stats[256 + t];
    if (t == 0)  Cs = stats[385];
    __syncthreads();
    const int b = blockIdx.x * 256 + t;
    float acc = Cs;
#pragma unroll 8
    for (int p = 0; p < NPATH; ++p)
        acc += p_ws[(size_t)p * BATCH + b] * coef[p];   // coalesced; p_ws lives in L2
    out[b] = 1.f / (1.f + __expf(-acc));
}

// ---------------------------------------------------------------------------
extern "C" void kernel_launch(void* const* d_in, const int* in_sizes, int n_in,
                              void* d_out, int out_size, void* d_ws, size_t ws_size,
                              hipStream_t stream) {
    const float* x     = (const float*)d_in[0];
    const float* W1    = (const float*)d_in[1];
    const float* W2    = (const float*)d_in[2];
    const float* gamma = (const float*)d_in[3];
    const float* beta  = (const float*)d_in[4];
    const float* Wd    = (const float*)d_in[5];
    const float* bd    = (const float*)d_in[6];
    float* out  = (float*)d_out;
    float* p_ws = (float*)d_ws;                                        // [P][B] col-major
    float* stats = (float*)((char*)d_ws + (size_t)NPATH * BATCH * sizeof(float));

    hipLaunchKernelGGL(init_kernel, dim3(1), dim3(1), 0, stream, stats);
    hipLaunchKernelGGL(pathway_mlp_kernel, dim3(NPATH, 8), dim3(128), 0, stream,
                       x, W1, W2, p_ws);
    hipLaunchKernelGGL(bn_stats_kernel, dim3(NPATH), dim3(256), 0, stream,
                       p_ws, gamma, beta, stats);
    hipLaunchKernelGGL(coef_kernel, dim3(1), dim3(128), 0, stream,
                       Wd, bd, gamma, beta, stats);
    hipLaunchKernelGGL(out_kernel, dim3(BATCH / 256), dim3(256), 0, stream,
                       p_ws, stats, out);
}
